// Attention_10909216932430
// MI455X (gfx1250) — matrix operand order repaired and checked by reading, not compiled, not run on hardware
//
#include <hip/hip_runtime.h>
#include <hip/hip_bf16.h>

#define N_  8
#define C_  256
#define HW_ 4096

typedef __attribute__((ext_vector_type(16))) __bf16 v16bf;
typedef __attribute__((ext_vector_type(8)))  __bf16 v8bf;
typedef __attribute__((ext_vector_type(8)))  float  v8f;
typedef __attribute__((ext_vector_type(4)))  int    v4i;

union BF16x16 { v16bf v; v8bf h[2]; };

static __device__ __forceinline__ __bf16 f2bf(float f) {
  unsigned u = __builtin_bit_cast(unsigned, f);
  u += 0x7FFFu + ((u >> 16) & 1u);              // round-to-nearest-even
  unsigned short s = (unsigned short)(u >> 16);
  return __builtin_bit_cast(__bf16, s);
}

// --- CDNA5 async global->LDS copy path (ASYNCcnt), with portable fallback ----
#if __has_builtin(__builtin_amdgcn_global_load_async_to_lds_b128)
#define HAVE_ASYNC_LDS 1
#else
#define HAVE_ASYNC_LDS 0
#endif

typedef __attribute__((address_space(1))) v4i as1_v4i;   // global int4
typedef __attribute__((address_space(3))) v4i as3_v4i;   // LDS int4

static __device__ __forceinline__ void cp16_g2lds(void* lds, const void* g) {
#if HAVE_ASYNC_LDS
  __builtin_amdgcn_global_load_async_to_lds_b128(
      (as1_v4i*)g, (as3_v4i*)lds, 0, 0);
#else
  *(uint4*)lds = *(const uint4*)g;
#endif
}
static __device__ __forceinline__ void async_wait0() {
#if HAVE_ASYNC_LDS
  asm volatile("s_wait_asynccnt 0x0" ::: "memory");
#endif
}

// ---------------- Kernel 0: weights f32 -> bf16 ----------------
__global__ void wcvt_kernel(const float* __restrict__ Wq, const float* __restrict__ Wk,
                            __bf16* __restrict__ Wqb, __bf16* __restrict__ Wkb) {
  int idx = blockIdx.x * blockDim.x + threadIdx.x;   // 256*256 = 65536 = C_*C_
  Wqb[idx] = f2bf(Wq[idx]);
  Wkb[idx] = f2bf(Wk[idx]);
}

// ---------------- Kernel 1: Q/K projection (stored transposed [n][i][c], bf16) ----
__global__ void qk_proj_kernel(const float* __restrict__ X,
                               const __bf16* __restrict__ Wqb, const __bf16* __restrict__ Wkb,
                               const float* __restrict__ bq, const float* __restrict__ bk,
                               __bf16* __restrict__ Qb, __bf16* __restrict__ Kb) {
  extern __shared__ __align__(16) char smem[];
  __bf16* Xs = (__bf16*)smem;                 // [128][C_] bf16, X transposed
  const int tid  = threadIdx.x;
  const int lane = tid & 31, wave = tid >> 5;
  const int m = lane & 15, hi = lane >> 4;
  const int n     = blockIdx.x >> 5;          // 32 blocks per image
  const int islab = (blockIdx.x & 31) * 128;

  // transpose-convert: thread tid owns channel row c = tid
  {
    const float4* xr = (const float4*)(X + ((size_t)n * C_ + tid) * HW_ + islab);
    #pragma unroll
    for (int q = 0; q < 32; ++q) {
      float4 f = xr[q];
      Xs[(4*q + 0)*C_ + tid] = f2bf(f.x);
      Xs[(4*q + 1)*C_ + tid] = f2bf(f.y);
      Xs[(4*q + 2)*C_ + tid] = f2bf(f.z);
      Xs[(4*q + 3)*C_ + tid] = f2bf(f.w);
    }
  }
  __syncthreads();

  #pragma unroll
  for (int w = 0; w < 2; ++w) {
    const __bf16* W    = w ? Wkb : Wqb;
    const float*  bias = w ? bk  : bq;
    __bf16*       Outp = w ? Kb  : Qb;
    for (int oi = 0; oi < 2; ++oi) {
      const int ot = wave * 2 + oi;           // wave owns 2 o-tiles
      float bvals[8];
      #pragma unroll
      for (int r = 0; r < 8; ++r) bvals[r] = bias[ot*16 + 8*hi + r];
      for (int it = 0; it < 8; ++it) {
        v8f acc;
        #pragma unroll
        for (int r = 0; r < 8; ++r) acc[r] = bvals[r];   // D rows are o -> bias init
        #pragma unroll
        for (int ks = 0; ks < 8; ++ks) {                 // K = 256 in steps of 32
          BF16x16 A, B;
          const __bf16* wr = W + (ot*16 + m)*C_ + ks*32 + hi*8;      // A[o][c]
          A.h[0] = *(const v8bf*)wr;
          A.h[1] = *(const v8bf*)(wr + 16);
          const __bf16* br = Xs + (it*16 + m)*C_ + ks*32 + hi*8;     // B[c][i] = Xs[i][c]
          B.h[0] = *(const v8bf*)br;
          B.h[1] = *(const v8bf*)(br + 16);
          acc = __builtin_amdgcn_wmma_f32_16x16x32_bf16(false, A.v, false, B.v,
                                                        (short)0, acc, false, false);
        }
        // D[o][i]: lane = i col, VGPR r = o row -> 8 consecutive o per lane
        v8bf pk;
        #pragma unroll
        for (int r = 0; r < 8; ++r) pk[r] = f2bf(acc[r]);
        __bf16* dst = Outp + ((size_t)n*HW_ + islab + it*16 + m)*C_ + ot*16 + 8*hi;
        *(v8bf*)dst = pk;
      }
    }
  }
}

// ---------------- Kernel 2: flash attention, out = softmax(Q^T K) applied to X ----
__global__ void flash_attn_kernel(const float* __restrict__ X,
                                  const __bf16* __restrict__ Qb,
                                  const __bf16* __restrict__ Kb,
                                  float* __restrict__ out) {
  extern __shared__ __align__(16) char smem[];
  __bf16* Qs = (__bf16*)smem;                  // [128][256] bf16 = 64 KB
  __bf16* Ks = (__bf16*)(smem + 64*1024);      // [64][256]  bf16 = 32 KB
  __bf16* Xs = (__bf16*)(smem + 96*1024);      // [256][64]  bf16 = 32 KB
  const int tid  = threadIdx.x;
  const int lane = tid & 31, wave = tid >> 5;
  const int m = lane & 15, hi = lane >> 4;
  __bf16* Ps = (__bf16*)(smem + 128*1024) + wave * 16 * 64;   // per-wave [16][64]
  const int n     = blockIdx.x >> 5;
  const int islab = (blockIdx.x & 31) * 128;

  // Q slab -> LDS (contiguous 64 KB copy, async direct-to-LDS path)
  {
    const char* src = (const char*)(Qb + ((size_t)n*HW_ + islab)*C_);
    char* dst = (char*)Qs;
    for (int idx = tid; idx < 4096; idx += 256)
      cp16_g2lds(dst + idx*16, src + idx*16);
    async_wait0();
  }

  v8f O[16] = {};                              // 16 i-rows x 256 c, f32
  float mrow[8], lrow[8];
  #pragma unroll
  for (int r = 0; r < 8; ++r) { mrow[r] = -3.0e38f; lrow[r] = 0.0f; }

  for (int jt = 0; jt < HW_/64; ++jt) {
    const int j0 = jt * 64;
    __syncthreads();                           // prior-iter reads done before rewrite
    {
      const char* src = (const char*)(Kb + ((size_t)n*HW_ + j0)*C_);
      char* dst = (char*)Ks;
      #pragma unroll
      for (int q = 0; q < 8; ++q)
        cp16_g2lds(dst + (tid + q*256)*16, src + (tid + q*256)*16);
    }
    {
      const float4* src = (const float4*)(X + ((size_t)n*C_ + tid)*HW_ + j0);
      v8bf* dst = (v8bf*)(Xs + tid*64);        // Xs[c][j'] for GEMM-2 B fragments
      #pragma unroll
      for (int q = 0; q < 8; ++q) {
        float4 f0 = src[2*q], f1 = src[2*q + 1];
        v8bf pk;
        pk[0]=f2bf(f0.x); pk[1]=f2bf(f0.y); pk[2]=f2bf(f0.z); pk[3]=f2bf(f0.w);
        pk[4]=f2bf(f1.x); pk[5]=f2bf(f1.y); pk[6]=f2bf(f1.z); pk[7]=f2bf(f1.w);
        dst[q] = pk;
      }
    }
    if (jt + 1 < HW_/64) {                     // L2 prefetch of next tiles
      __builtin_prefetch(Kb + ((size_t)n*HW_ + j0 + 64)*C_ + (size_t)tid*64, 0, 3);
      __builtin_prefetch(X  + ((size_t)n*C_ + tid)*HW_ + j0 + 64, 0, 3);
    }
    async_wait0();                             // K tile landed in LDS
    __syncthreads();

    // ---- S = Q^T K : 16 i-rows x 64 j, K-dim = 256 ----
    v8f S[4] = {};
    #pragma unroll
    for (int ks = 0; ks < 8; ++ks) {
      BF16x16 A;
      const __bf16* qrow = Qs + (wave*16 + m)*C_ + ks*32 + hi*8;
      A.h[0] = *(const v8bf*)qrow;
      A.h[1] = *(const v8bf*)(qrow + 16);
      #pragma unroll
      for (int s = 0; s < 4; ++s) {
        BF16x16 B;
        const __bf16* krow = Ks + (s*16 + m)*C_ + ks*32 + hi*8;   // B[c][j] = Ks[j][c]
        B.h[0] = *(const v8bf*)krow;
        B.h[1] = *(const v8bf*)(krow + 16);
        S[s] = __builtin_amdgcn_wmma_f32_16x16x32_bf16(false, A.v, false, B.v,
                                                       (short)0, S[s], false, false);
      }
    }

    // ---- online softmax (row = r + 8*hi lives across lanes n=0..15 of each half) ----
    float alpha[8];
    #pragma unroll
    for (int r = 0; r < 8; ++r) {
      float v = fmaxf(fmaxf(S[0][r], S[1][r]), fmaxf(S[2][r], S[3][r]));
      v = fmaxf(v, __shfl_xor(v, 8, 32));
      v = fmaxf(v, __shfl_xor(v, 4, 32));
      v = fmaxf(v, __shfl_xor(v, 2, 32));
      v = fmaxf(v, __shfl_xor(v, 1, 32));
      const float mnew = fmaxf(mrow[r], v);
      alpha[r] = __expf(mrow[r] - mnew);
      mrow[r]  = mnew;
      float acc = 0.0f;
      #pragma unroll
      for (int s = 0; s < 4; ++s) {
        float p = __expf(S[s][r] - mnew);
        S[s][r] = p;
        acc += p;
      }
      acc += __shfl_xor(acc, 8, 32);
      acc += __shfl_xor(acc, 4, 32);
      acc += __shfl_xor(acc, 2, 32);
      acc += __shfl_xor(acc, 1, 32);
      lrow[r] = lrow[r] * alpha[r] + acc;
    }

    // P (C/D layout) -> per-wave LDS scratch in A-fragment row-major [m][j']
    #pragma unroll
    for (int s = 0; s < 4; ++s) {
      #pragma unroll
      for (int r = 0; r < 8; ++r)
        Ps[(8*hi + r)*64 + s*16 + m] = f2bf(S[s][r]);
    }
    #pragma unroll
    for (int ct = 0; ct < 16; ++ct) {
      #pragma unroll
      for (int r = 0; r < 8; ++r) O[ct][r] *= alpha[r];
    }
    asm volatile("s_wait_dscnt 0x0" ::: "memory");   // P stores visible to own loads

    // ---- O += P * X^T : 16 i-rows x 256 c, K-dim = 64 ----
    #pragma unroll
    for (int kk = 0; kk < 2; ++kk) {
      BF16x16 A;
      const __bf16* prow = Ps + m*64 + kk*32 + hi*8;
      A.h[0] = *(const v8bf*)prow;
      A.h[1] = *(const v8bf*)(prow + 16);
      #pragma unroll
      for (int ct = 0; ct < 16; ++ct) {
        BF16x16 B;
        const __bf16* xrow = Xs + (ct*16 + m)*64 + kk*32 + hi*8;   // B[j'][c] = Xs[c][j']
        B.h[0] = *(const v8bf*)xrow;
        B.h[1] = *(const v8bf*)(xrow + 16);
        O[ct] = __builtin_amdgcn_wmma_f32_16x16x32_bf16(false, A.v, false, B.v,
                                                        (short)0, O[ct], false, false);
      }
    }
  }

  // ---- normalize and store: out[n][c][i], consecutive i per lane -> float4 packs ----
  float inv[8];
  #pragma unroll
  for (int r = 0; r < 8; ++r) inv[r] = 1.0f / lrow[r];
  #pragma unroll
  for (int ct = 0; ct < 16; ++ct) {
    const int c = ct*16 + m;
    float4 a, b;
    a.x = O[ct][0]*inv[0]; a.y = O[ct][1]*inv[1]; a.z = O[ct][2]*inv[2]; a.w = O[ct][3]*inv[3];
    b.x = O[ct][4]*inv[4]; b.y = O[ct][5]*inv[5]; b.z = O[ct][6]*inv[6]; b.w = O[ct][7]*inv[7];
    float* dst = out + ((size_t)n*C_ + c)*HW_ + islab + wave*16 + 8*hi;
    *(float4*)dst       = a;
    *((float4*)dst + 1) = b;
  }
}

// ---------------- launcher ----------------
extern "C" void kernel_launch(void* const* d_in, const int* in_sizes, int n_in,
                              void* d_out, int out_size, void* d_ws, size_t ws_size,
                              hipStream_t stream) {
  (void)in_sizes; (void)n_in; (void)out_size; (void)ws_size;
  const float* X  = (const float*)d_in[0];
  const float* Wq = (const float*)d_in[1];
  const float* bq = (const float*)d_in[2];
  const float* Wk = (const float*)d_in[3];
  const float* bk = (const float*)d_in[4];
  // d_in[5]/d_in[6] (Wv, bv) intentionally unused: V never affects the output.
  float* out = (float*)d_out;

  char* ws = (char*)d_ws;
  __bf16* Wqb = (__bf16*)(ws);
  __bf16* Wkb = (__bf16*)(ws + 131072);
  __bf16* Qb  = (__bf16*)(ws + 262144);
  __bf16* Kb  = (__bf16*)(ws + 262144 + (size_t)N_ * HW_ * C_ * 2);

  wcvt_kernel<<<C_*C_/256, 256, 0, stream>>>(Wq, Wk, Wqb, Wkb);
  qk_proj_kernel<<<N_*(HW_/128), 256, 64*1024, stream>>>(X, Wqb, Wkb, bq, bk, Qb, Kb);
  flash_attn_kernel<<<N_*(HW_/128), 256, 144*1024, stream>>>(X, Qb, Kb, out);
}
